// GNN_74577812128001
// MI455X (gfx1250) — matrix operand order repaired and checked by reading, not compiled, not run on hardware
//
#include <hip/hip_runtime.h>

typedef __attribute__((ext_vector_type(16))) _Float16 v16h;
typedef __attribute__((ext_vector_type(8)))  _Float16 v8h;
typedef __attribute__((ext_vector_type(8)))  float    v8f;

#define TPB 256

__device__ __forceinline__ float fast_sigmoid(float x) {
    // v_exp_f32 + v_rcp_f32 instead of the IEEE div expansion
    return __builtin_amdgcn_rcpf(1.f + __expf(-x));
}

// ---------------------------------------------------------------- utilities

__global__ void zero_f32_kernel(float* __restrict__ p, long long n) {
    long long i  = (long long)blockIdx.x * blockDim.x + threadIdx.x;
    long long st = (long long)gridDim.x * blockDim.x;
    for (; i < n; i += st) p[i] = 0.f;
}

__global__ void f32_to_f16_kernel(const float* __restrict__ s,
                                  _Float16* __restrict__ d, long long n) {
    long long i  = (long long)blockIdx.x * blockDim.x + threadIdx.x;
    long long st = (long long)gridDim.x * blockDim.x;
    for (; i < n; i += st) d[i] = (_Float16)s[i];
}

// h = relu(x @ hp_w + hp_b)   x:[N,2]  w:[2,64]
__global__ void node_proj_kernel(const float* __restrict__ x,
                                 const float* __restrict__ w,
                                 const float* __restrict__ b,
                                 float* __restrict__ h, long long total) {
    long long i  = (long long)blockIdx.x * blockDim.x + threadIdx.x;
    long long st = (long long)gridDim.x * blockDim.x;
    for (; i < total; i += st) {
        long long node = i >> 6; int d = (int)(i & 63);
        float v = fmaf(x[node * 2 + 1], w[64 + d], fmaf(x[node * 2], w[d], b[d]));
        h[i] = fmaxf(v, 0.f);
    }
}

// e = relu(edge_attr @ ep_w + ep_b)   edge_attr:[E,1]  w:[1,64]
__global__ void edge_proj_kernel(const float* __restrict__ ea,
                                 const float* __restrict__ w,
                                 const float* __restrict__ b,
                                 float* __restrict__ e, long long total) {
    long long i  = (long long)blockIdx.x * blockDim.x + threadIdx.x;
    long long st = (long long)gridDim.x * blockDim.x;
    for (; i < total; i += st) {
        long long j = i >> 6; int d = (int)(i & 63);
        float v = fmaf(ea[j], w[d], b[d]);
        e[i] = fmaxf(v, 0.f);
    }
}

__global__ void degree_kernel(const int* __restrict__ dst,
                              float* __restrict__ deg, long long E_) {
    long long i  = (long long)blockIdx.x * blockDim.x + threadIdx.x;
    long long st = (long long)gridDim.x * blockDim.x;
    for (; i < E_; i += st) atomicAdd(&deg[dst[i]], 1.f);
}

// ---------------------------------------------------------------- WMMA GEMM
// out[M,64] = act(A[M,64] @ W[64,64] + bias)   A,W in f16, out f32.
// ACT: 0=none, 2=silu (compile-time)
template <int ACT>
__global__ __launch_bounds__(TPB)
void gemm64_kernel(const _Float16* __restrict__ A,
                   const _Float16* __restrict__ W,
                   const float* __restrict__ bias,
                   float* __restrict__ out,
                   int M) {
    // Weight tile staged TRANSPOSED in LDS: wlT[n*64 + k] = W[k*64 + n].
    // B fragment (col n, K=kb..kb+7 and kb+16..kb+23) is then 2 contiguous ds_load_b128.
    __shared__ _Float16 wlT[64 * 64];
    {
#pragma unroll
        for (int i = threadIdx.x; i < 4096; i += TPB) {
            int k = i >> 6, n = i & 63;          // coalesced global read
            wlT[n * 64 + k] = W[i];              // one-time scatter into LDS
        }
    }
    __syncthreads();

    const int wave     = threadIdx.x >> 5;
    const int lane     = threadIdx.x & 31;
    const int halfLane = lane & 15;
    const int laneHi   = lane >> 4;          // 0 or 1
    const int r0 = blockIdx.x * 32 + (wave >> 2) * 16;   // row tile
    const int c0 = (wave & 3) * 16;                      // col tile
    if (r0 >= M) return;                                 // wave-uniform exit

    v8f acc = {};
#pragma unroll
    for (int ks = 0; ks < 64; ks += 32) {
        // A fragment: lane<16 -> K base ks; lane>=16 -> +8.
        // elems 0..7 = K kbase..kbase+7 ; elems 8..15 = K kbase+16..kbase+23
        int arow = r0 + halfLane;
        if (arow >= M) arow = M - 1;         // clamp (tail rows not stored)
        const _Float16* ap = A + (long long)arow * 64 + ks + laneHi * 8;
        v8h alo = *(const v8h*)ap;
        v8h ahi = *(const v8h*)(ap + 16);
        v16h a = __builtin_shufflevector(alo, ahi,
                 0,1,2,3,4,5,6,7,8,9,10,11,12,13,14,15);
        // B fragment: same K order as A (kb+e, kb+16+e with kb = ks + laneHi*8); N = c0+halfLane
        const int kb = ks + laneHi * 8;
        const _Float16* bp = wlT + (c0 + halfLane) * 64 + kb;
        v8h blo = *(const v8h*)bp;           // ds_load_b128
        v8h bhi = *(const v8h*)(bp + 16);    // ds_load_b128
        v16h b = __builtin_shufflevector(blo, bhi,
                 0,1,2,3,4,5,6,7,8,9,10,11,12,13,14,15);
        acc = __builtin_amdgcn_wmma_f32_16x16x32_f16(
                  false, a, false, b, (short)0, acc, false, false);
    }

    // Epilogue: bias + activation in registers, then stores with immediate
    // offsets. Tail bounds-check hoisted to one wave-uniform branch.
    const float bv = bias[c0 + halfLane];
    float res[8];
#pragma unroll
    for (int i = 0; i < 8; ++i) {
        float v = acc[i] + bv;
        if (ACT == 1) v = fmaxf(v, 0.f);
        if (ACT == 2) v = v * fast_sigmoid(v);
        res[i] = v;
    }
    const int mBase = r0 + laneHi * 8;       // C/D layout: VGPR i -> row mBase+i
    float* op = out + (long long)mBase * 64 + c0 + halfLane;
    if (r0 + 16 <= M) {                      // full tile: straight-line stores
#pragma unroll
        for (int i = 0; i < 8; ++i) op[i * 64] = res[i];
    } else {                                 // tail tile only
#pragma unroll
        for (int i = 0; i < 8; ++i)
            if (mBase + i < M) op[i * 64] = res[i];
    }
}

// ------------------------------------------------- fused edge gather kernel
// per element (edge j, channel d):
//   gate = sigmoid(e_in); atomicAdd(agg[dst], gate*Vh[src])
//   Ae := Ae + Bh[dst] + Ch[src]   (in place)  + e-BN sum/sumsq reduction
__global__ void edge_gather_kernel(const float* __restrict__ e,
                                   const float* __restrict__ Vh,
                                   const float* __restrict__ Bh,
                                   const float* __restrict__ Ch,
                                   float* __restrict__ Ae,
                                   float* __restrict__ agg,
                                   const int* __restrict__ dst,
                                   const int* __restrict__ src,
                                   float* __restrict__ sumOut,
                                   float* __restrict__ sumsqOut,
                                   long long total) {
    long long i  = (long long)blockIdx.x * blockDim.x + threadIdx.x;
    long long st = (long long)gridDim.x * blockDim.x;
    float s = 0.f, s2 = 0.f;
    for (; i < total; i += st) {
        long long j = i >> 6; int d = (int)(i & 63);
        int dj = dst[j], sj = src[j];
        float gate = fast_sigmoid(e[i]);
        float msg  = gate * Vh[(long long)sj * 64 + d];
        atomicAdd(&agg[(long long)dj * 64 + d], msg);
        float te = Ae[i] + Bh[(long long)dj * 64 + d] + Ch[(long long)sj * 64 + d];
        Ae[i] = te;
        s += te; s2 += te * te;
    }
    __shared__ float ls[TPB], ls2[TPB];
    ls[threadIdx.x] = s; ls2[threadIdx.x] = s2;
    __syncthreads();
    if (threadIdx.x < 64) {   // channel = tid&63 is invariant along the stride
        float a  = ls[threadIdx.x]  + ls[threadIdx.x + 64]  + ls[threadIdx.x + 128]  + ls[threadIdx.x + 192];
        float a2 = ls2[threadIdx.x] + ls2[threadIdx.x + 64] + ls2[threadIdx.x + 128] + ls2[threadIdx.x + 192];
        atomicAdd(&sumOut[threadIdx.x], a);
        atomicAdd(&sumsqOut[threadIdx.x], a2);
    }
}

// t = Uh + agg/max(deg,1)  (in place into Uh)  + h-BN sum/sumsq reduction
__global__ void node_bn_kernel(float* __restrict__ Uh,
                               const float* __restrict__ agg,
                               const float* __restrict__ deg,
                               float* __restrict__ sumOut,
                               float* __restrict__ sumsqOut,
                               long long total) {
    long long i  = (long long)blockIdx.x * blockDim.x + threadIdx.x;
    long long st = (long long)gridDim.x * blockDim.x;
    float s = 0.f, s2 = 0.f;
    for (; i < total; i += st) {
        float t = Uh[i] + agg[i] * __builtin_amdgcn_rcpf(fmaxf(deg[i >> 6], 1.f));
        Uh[i] = t;
        s += t; s2 += t * t;
    }
    __shared__ float ls[TPB], ls2[TPB];
    ls[threadIdx.x] = s; ls2[threadIdx.x] = s2;
    __syncthreads();
    if (threadIdx.x < 64) {
        float a  = ls[threadIdx.x]  + ls[threadIdx.x + 64]  + ls[threadIdx.x + 128]  + ls[threadIdx.x + 192];
        float a2 = ls2[threadIdx.x] + ls2[threadIdx.x + 64] + ls2[threadIdx.x + 128] + ls2[threadIdx.x + 192];
        atomicAdd(&sumOut[threadIdx.x], a);
        atomicAdd(&sumsqOut[threadIdx.x], a2);
    }
}

// stats layout (floats): [0]hsum [64]hsumsq [128]esum [192]esumsq
//                        [256]hmean [320]hinv [384]emean [448]einv
__global__ void finalize_stats_kernel(float* __restrict__ stats, float nN, float nE) {
    int t = threadIdx.x;           // 128 threads
    if (t < 64) {
        float m = stats[t] / nN;
        float v = stats[64 + t] / nN - m * m;
        stats[256 + t] = m;
        stats[320 + t] = rsqrtf(v + 1e-5f);
    } else {
        int d = t - 64;
        float m = stats[128 + d] / nE;
        float v = stats[192 + d] / nE - m * m;
        stats[384 + d] = m;
        stats[448 + d] = rsqrtf(v + 1e-5f);
    }
}

// io += relu((t - mean)*inv*gamma + beta)
__global__ void apply_bn_kernel(float* __restrict__ io,
                                const float* __restrict__ t,
                                const float* __restrict__ mean,
                                const float* __restrict__ inv,
                                const float* __restrict__ gamma,
                                const float* __restrict__ beta,
                                long long total) {
    long long i  = (long long)blockIdx.x * blockDim.x + threadIdx.x;
    long long st = (long long)gridDim.x * blockDim.x;
    for (; i < total; i += st) {
        int d = (int)(i & 63);
        float v = (t[i] - mean[d]) * inv[d] * gamma[d] + beta[d];
        io[i] += fmaxf(v, 0.f);
    }
}

// out[j] = sigmoid(sum_d z[j,d]*w[d] + b)
__global__ void out_kernel(const float* __restrict__ z,
                           const float* __restrict__ w,
                           const float* __restrict__ b,
                           float* __restrict__ out, long long E_) {
    long long j  = (long long)blockIdx.x * blockDim.x + threadIdx.x;
    long long st = (long long)gridDim.x * blockDim.x;
    for (; j < E_; j += st) {
        const float4* zp = (const float4*)(z + j * 64);
        const float4* wp = (const float4*)w;
        float s = b[0];
#pragma unroll
        for (int q = 0; q < 16; ++q) {
            float4 zv = zp[q], wv = wp[q];
            s = fmaf(zv.x, wv.x, s);
            s = fmaf(zv.y, wv.y, s);
            s = fmaf(zv.z, wv.z, s);
            s = fmaf(zv.w, wv.w, s);
        }
        out[j] = fast_sigmoid(s);
    }
}

// ---------------------------------------------------------------- launcher

extern "C" void kernel_launch(void* const* d_in, const int* in_sizes, int n_in,
                              void* d_out, int out_size, void* d_ws, size_t ws_size,
                              hipStream_t stream) {
    const float* x     = (const float*)d_in[0];
    const float* eattr = (const float*)d_in[1];
    const int*   eidx  = (const int*)  d_in[2];
    const float* hp_w  = (const float*)d_in[3];
    const float* hp_b  = (const float*)d_in[4];
    const float* ep_w  = (const float*)d_in[5];
    const float* ep_b  = (const float*)d_in[6];
    const float* Uw = (const float*)d_in[7];   const float* Ub = (const float*)d_in[8];
    const float* Vw = (const float*)d_in[9];   const float* Vb = (const float*)d_in[10];
    const float* Aw = (const float*)d_in[11];  const float* Ab = (const float*)d_in[12];
    const float* Bw = (const float*)d_in[13];  const float* Bb = (const float*)d_in[14];
    const float* Cw = (const float*)d_in[15];  const float* Cb = (const float*)d_in[16];
    const float* hg = (const float*)d_in[17];  const float* hb = (const float*)d_in[18];
    const float* eg = (const float*)d_in[19];  const float* eb = (const float*)d_in[20];
    const float* f1w = (const float*)d_in[21]; const float* f1b = (const float*)d_in[22];
    const float* f2w = (const float*)d_in[23]; const float* f2b = (const float*)d_in[24];
    const float* f3w = (const float*)d_in[25]; const float* f3b = (const float*)d_in[26];
    float* out = (float*)d_out;
    (void)n_in; (void)out_size; (void)ws_size;

    const long long N = in_sizes[0] / 2;
    const long long E = in_sizes[1];
    const int Ldep = in_sizes[7] / (64 * 64);
    const int* dstI = eidx;        // edge_index[0]
    const int* srcI = eidx + E;    // edge_index[1]

    // ---- workspace carve (~596 MB) ----
    char* base = (char*)d_ws;
    size_t off = 0;
    auto carve = [&](size_t bytes) -> char* {
        char* p = base + off;
        off += (bytes + 255) & ~(size_t)255;
        return p;
    };
    float*    eF   = (float*)   carve((size_t)E * 64 * 4);
    float*    AeF  = (float*)   carve((size_t)E * 64 * 4);
    _Float16* eH   = (_Float16*)carve((size_t)E * 64 * 2);
    float*    hF   = (float*)   carve((size_t)N * 64 * 4);
    _Float16* hH   = (_Float16*)carve((size_t)N * 64 * 2);
    float*    UhF  = (float*)   carve((size_t)N * 64 * 4);
    float*    VhF  = (float*)   carve((size_t)N * 64 * 4);
    float*    BhF  = (float*)   carve((size_t)N * 64 * 4);
    float*    ChF  = (float*)   carve((size_t)N * 64 * 4);
    float*    aggF = (float*)   carve((size_t)N * 64 * 4);
    float*    degF = (float*)   carve((size_t)N * 4);
    _Float16* UwH  = (_Float16*)carve((size_t)Ldep * 4096 * 2);
    _Float16* VwH  = (_Float16*)carve((size_t)Ldep * 4096 * 2);
    _Float16* AwH  = (_Float16*)carve((size_t)Ldep * 4096 * 2);
    _Float16* BwH  = (_Float16*)carve((size_t)Ldep * 4096 * 2);
    _Float16* CwH  = (_Float16*)carve((size_t)Ldep * 4096 * 2);
    _Float16* f1wH = (_Float16*)carve(4096 * 2);
    _Float16* f2wH = (_Float16*)carve(4096 * 2);
    float*    stats= (float*)   carve(512 * 4);

    auto gridFor = [](long long total) -> unsigned {
        long long b = (total + TPB - 1) / TPB;
        if (b > 16384) b = 16384;
        if (b < 1) b = 1;
        return (unsigned)b;
    };
    const unsigned gNE = gridFor(N * 64);
    const unsigned gEE = gridFor(E * 64);
    const unsigned gemmN = (unsigned)((N + 31) / 32);
    const unsigned gemmE = (unsigned)((E + 31) / 32);
    const long long LW = (long long)Ldep * 4096;

    // weight conversions to f16
    f32_to_f16_kernel<<<gridFor(LW), TPB, 0, stream>>>(Uw, UwH, LW);
    f32_to_f16_kernel<<<gridFor(LW), TPB, 0, stream>>>(Vw, VwH, LW);
    f32_to_f16_kernel<<<gridFor(LW), TPB, 0, stream>>>(Aw, AwH, LW);
    f32_to_f16_kernel<<<gridFor(LW), TPB, 0, stream>>>(Bw, BwH, LW);
    f32_to_f16_kernel<<<gridFor(LW), TPB, 0, stream>>>(Cw, CwH, LW);
    f32_to_f16_kernel<<<gridFor(4096), TPB, 0, stream>>>(f1w, f1wH, 4096);
    f32_to_f16_kernel<<<gridFor(4096), TPB, 0, stream>>>(f2w, f2wH, 4096);

    // input projections
    node_proj_kernel<<<gNE, TPB, 0, stream>>>(x, hp_w, hp_b, hF, N * 64);
    edge_proj_kernel<<<gEE, TPB, 0, stream>>>(eattr, ep_w, ep_b, eF, E * 64);

    // destination degree (same for every layer)
    zero_f32_kernel<<<gridFor(N), TPB, 0, stream>>>(degF, N);
    degree_kernel<<<gridFor(E), TPB, 0, stream>>>(dstI, degF, E);

    for (int l = 0; l < Ldep; ++l) {
        f32_to_f16_kernel<<<gNE, TPB, 0, stream>>>(hF, hH, N * 64);
        f32_to_f16_kernel<<<gEE, TPB, 0, stream>>>(eF, eH, E * 64);

        gemm64_kernel<0><<<gemmN, TPB, 0, stream>>>(hH, UwH + l * 4096, Ub + l * 64, UhF, (int)N);
        gemm64_kernel<0><<<gemmN, TPB, 0, stream>>>(hH, VwH + l * 4096, Vb + l * 64, VhF, (int)N);
        gemm64_kernel<0><<<gemmN, TPB, 0, stream>>>(hH, BwH + l * 4096, Bb + l * 64, BhF, (int)N);
        gemm64_kernel<0><<<gemmN, TPB, 0, stream>>>(hH, CwH + l * 4096, Cb + l * 64, ChF, (int)N);
        gemm64_kernel<0><<<gemmE, TPB, 0, stream>>>(eH, AwH + l * 4096, Ab + l * 64, AeF, (int)E);

        zero_f32_kernel<<<gNE, TPB, 0, stream>>>(aggF, N * 64);
        zero_f32_kernel<<<1, TPB, 0, stream>>>(stats, 256);

        edge_gather_kernel<<<gEE, TPB, 0, stream>>>(eF, VhF, BhF, ChF, AeF, aggF,
                                                    dstI, srcI,
                                                    stats + 128, stats + 192, E * 64);
        node_bn_kernel<<<gNE, TPB, 0, stream>>>(UhF, aggF, degF,
                                                stats + 0, stats + 64, N * 64);
        finalize_stats_kernel<<<1, 128, 0, stream>>>(stats, (float)N, (float)E);
        apply_bn_kernel<<<gNE, TPB, 0, stream>>>(hF, UhF, stats + 256, stats + 320,
                                                 hg + l * 64, hb + l * 64, N * 64);
        apply_bn_kernel<<<gEE, TPB, 0, stream>>>(eF, AeF, stats + 384, stats + 448,
                                                 eg + l * 64, eb + l * 64, E * 64);
    }

    // feature extractor MLP: silu, silu, sigmoid-dot
    f32_to_f16_kernel<<<gEE, TPB, 0, stream>>>(eF, eH, E * 64);
    gemm64_kernel<2><<<gemmE, TPB, 0, stream>>>(eH, f1wH, f1b, AeF, (int)E);
    f32_to_f16_kernel<<<gEE, TPB, 0, stream>>>(AeF, eH, E * 64);
    gemm64_kernel<2><<<gemmE, TPB, 0, stream>>>(eH, f2wH, f2b, eF, (int)E);
    out_kernel<<<gridFor(E), TPB, 0, stream>>>(eF, f3w, f3b, out, E);
}
